// LSTMClassifier_31885837205940
// MI455X (gfx1250) — matrix operand order repaired and checked by reading, not compiled, not run on hardware
//
#include <hip/hip_runtime.h>

// LSTM classifier fused kernel for gfx1250 (MI455X), wave32 + WMMA.
//
//   gates(t)^T[128 x 16] = W_hh[128x32] * h^T  +  G[128x32] * onehot(tok)
// with G = W_ih @ emb^T + (b_ih + b_hh), vocab padded 24->32.
// Both products are chained v_wmma_f32_16x16x32_f16; W_hh and G live as
// f16 A-fragments in VGPRs for the whole scan. Rows of the sigmoid gates
// (i,f,o) are pre-scaled by 0.5 so sigmoid(x) = 0.5 + 0.5*tanh(d) needs no
// input multiply. One-hot B rows come from a small LDS table (DS pipe
// co-executes with VALU). Activations use v_tanh_f32.

#define VOCAB 24
#define E_DIM 16
#define H_DIM 32
#define T_LEN 128
#define B_TOT 16384
#define WAVES 4
#define ROWS_PER_WAVE 16
#define ROWS_PER_BLOCK (WAVES * ROWS_PER_WAVE)
#define NTHREADS (WAVES * 32)

typedef __attribute__((ext_vector_type(16))) _Float16 v16h;
typedef __attribute__((ext_vector_type(8)))  float    v8f;
typedef __attribute__((ext_vector_type(8)))  int      v8i;
typedef __attribute__((ext_vector_type(4)))  float    v4f;
typedef __attribute__((ext_vector_type(4)))  int      v4i;

#if __has_builtin(__builtin_amdgcn_tanhf)
__device__ __forceinline__ float fast_tanh(float x) {
  return __builtin_amdgcn_tanhf(x);                       // v_tanh_f32
}
// input pre-scaled by 0.5 in the weights: sigmoid(2x) = 0.5 + 0.5*tanh(x)
__device__ __forceinline__ float fast_sigmoid_pre(float x) {
  return fmaf(0.5f, __builtin_amdgcn_tanhf(x), 0.5f);
}
#else
__device__ __forceinline__ float fast_tanh(float x) {
  return 1.0f - 2.0f * __builtin_amdgcn_rcpf(1.0f + __expf(2.0f * x));
}
__device__ __forceinline__ float fast_sigmoid_pre(float x) {
  return __builtin_amdgcn_rcpf(1.0f + __expf(-2.0f * x));
}
#endif

__device__ __forceinline__ int   f2i(float x) { return __builtin_bit_cast(int, x); }
__device__ __forceinline__ float i2f(int x)   { return __builtin_bit_cast(float, x); }

// ds_swizzle SWAPX16: group-of-32 mode, xor=0x10, or=0, and=0x1f -> 0x401F
#define SWZ_SWAPX16 0x401F

__global__ __launch_bounds__(NTHREADS) void lstm_fused(
    const int*   __restrict__ x,     // [B, T]
    const float* __restrict__ emb,   // [24, 16]
    const float* __restrict__ W_ih,  // [128, 16]
    const float* __restrict__ W_hh,  // [128, 32]
    const float* __restrict__ b_ih,  // [128]
    const float* __restrict__ b_hh,  // [128]
    const float* __restrict__ fc_w,  // [2, 32]
    const float* __restrict__ fc_b,  // [2]
    float*       __restrict__ out)   // [B, 2]
{
  __shared__ int   toks[WAVES][T_LEN][17];   // pad 17 -> conflict-free
  __shared__ int   onehot_tbl[VOCAB][16];    // packed f16x2 one-hot rows
  __shared__ float G_lds[128][36];           // pad 36 -> conflict-free, 16B aligned

  const int tid     = threadIdx.x;
  const int wave    = tid >> 5;
  const int lane    = tid & 31;
  const int lane16  = lane & 15;
  const int halfsel = lane >> 4;   // 0: lanes 0-15, 1: lanes 16-31
  const int rowBase = blockIdx.x * ROWS_PER_BLOCK + wave * ROWS_PER_WAVE;

  // ---------------- stage this wave's tokens -------------------------------------
  for (int r = 0; r < ROWS_PER_WAVE; ++r) {
    const int* xr = x + (size_t)(rowBase + r) * T_LEN;
    #pragma unroll
    for (int q = 0; q < 4; ++q) {
      toks[wave][lane + 32 * q][r] = xr[lane + 32 * q];  // coalesced 128B loads
    }
  }

  // ---------------- one-hot table: row tok -> 16 dwords of packed f16 pairs ------
  for (int i = tid; i < VOCAB * 16; i += NTHREADS) {
    const int row = i >> 4, vv = i & 15;   // dword vv covers K = {2vv, 2vv+1}
    onehot_tbl[row][vv] = ((2 * vv == row) ? 0x3C00 : 0) |
                          ((2 * vv + 1 == row) ? 0x3C000000 : 0);
  }

  // ---------------- G = W_ih @ emb^T + bias, one gate-row per thread -------------
  {
    const v4f* wih = (const v4f*)(W_ih + (size_t)tid * E_DIM);
    const v4f w0 = wih[0], w1 = wih[1], w2 = wih[2], w3 = wih[3];
    const float bias = b_ih[tid] + b_hh[tid];
    #pragma unroll
    for (int vv = 0; vv < VOCAB; ++vv) {
      const v4f* er = (const v4f*)(emb + (size_t)vv * E_DIM);
      const v4f e0 = er[0], e1 = er[1], e2 = er[2], e3 = er[3];
      float val = bias;
      #pragma unroll
      for (int j = 0; j < 4; ++j)
        val += w0[j] * e0[j] + w1[j] * e1[j] + w2[j] * e2[j] + w3[j] * e3[j];
      G_lds[tid][vv] = val;
    }
    #pragma unroll
    for (int vv = VOCAB; vv < 32; ++vv) G_lds[tid][vv] = 0.0f;
  }
  __syncthreads();

  // ---------------- persistent A fragments (f16), built once ---------------------
  // A fragments (contiguous k order chosen here, matches B): lane row M = 16m + lane16;
// K slots: K = s + 16*halfsel.
  // Tiles 0..7 = gates [i,i,f,f,g,g,o,o]; sigmoid gate rows pre-scaled by 0.5.
  v16h a_whh[8], a_g[8];
  #pragma unroll
  for (int m = 0; m < 8; ++m) {
    const int g = 16 * m + lane16;       // gate row (PyTorch i,f,g,o order)
    const float sc = (m == 4 || m == 5) ? 1.0f : 0.5f;   // g-gate stays unscaled

    const v4f* wr = (const v4f*)(W_hh + (size_t)g * H_DIM);
    const v4f r0 = wr[4 * halfsel + 0];
    const v4f r1 = wr[4 * halfsel + 1];
    const v4f r2 = wr[4 * halfsel + 2];
    const v4f r3 = wr[4 * halfsel + 3];

    const float* grow = &G_lds[g][0];
    const v4f q0 = *(const v4f*)(grow + 16 * halfsel);
    const v4f q1 = *(const v4f*)(grow + 16 * halfsel + 4);
    const v4f q2 = *(const v4f*)(grow + 16 * halfsel + 8);
    const v4f q3 = *(const v4f*)(grow + 16 * halfsel + 12);

    v16h aw, ag;
    #pragma unroll
    for (int e = 0; e < 4; ++e) {
      aw[e]      = (_Float16)(sc * r0[e]);
      aw[4 + e]  = (_Float16)(sc * r1[e]);
      aw[8 + e]  = (_Float16)(sc * r2[e]);
      aw[12 + e] = (_Float16)(sc * r3[e]);
      ag[e]      = (_Float16)(sc * q0[e]);
      ag[4 + e]  = (_Float16)(sc * q1[e]);
      ag[8 + e]  = (_Float16)(sc * q2[e]);
      ag[12 + e] = (_Float16)(sc * q3[e]);
    }
    a_whh[m] = aw;
    a_g[m]   = ag;
  }

  // ---------------- recurrent state ----------------------------------------------
  // D-tile layout: lane holds gates[16m + v + 8*halfsel][batch=lane16].
  // hidden index j = v + 8*halfsel + 16*mm
  float hst[2][8], cst[2][8];
  #pragma unroll
  for (int v = 0; v < 8; ++v) { hst[0][v] = hst[1][v] = 0.0f; cst[0][v] = cst[1][v] = 0.0f; }

  // ---------------- time scan -----------------------------------------------------
  #pragma unroll 2
  for (int t = 0; t < T_LEN; ++t) {
    const int tok = toks[wave][t][lane16];   // token of this lane's batch column

    // one-hot B fragment from LDS table: B layout K(s) = s + 16*halfsel
    const v4i o0 = *(const v4i*)(&onehot_tbl[tok][8 * halfsel]);
    const v4i o1 = *(const v4i*)(&onehot_tbl[tok][8 * halfsel + 4]);
    v8i bo_pk;
    #pragma unroll
    for (int k = 0; k < 4; ++k) { bo_pk[k] = o0[k]; bo_pk[4 + k] = o1[k]; }
    const v16h bo = __builtin_bit_cast(v16h, bo_pk);

    // h B fragment: pack to f16, swap half-wave partner's chunk via ds_swizzle
    int lo[4], hi[4];
    #pragma unroll
    for (int k = 0; k < 4; ++k) {
      auto pl = __builtin_amdgcn_cvt_pkrtz(hst[0][2 * k], hst[0][2 * k + 1]);
      auto ph = __builtin_amdgcn_cvt_pkrtz(hst[1][2 * k], hst[1][2 * k + 1]);
      lo[k] = __builtin_bit_cast(int, pl);
      hi[k] = __builtin_bit_cast(int, ph);
    }
    v8i bh_pk;
    #pragma unroll
    for (int k = 0; k < 4; ++k) {
      const int sent = halfsel ? lo[k] : hi[k];                 // what partner needs
      const int recv = __builtin_amdgcn_ds_swizzle(sent, SWZ_SWAPX16);
      bh_pk[k]     = halfsel ? recv  : lo[k];
      bh_pk[4 + k] = halfsel ? hi[k] : recv;
    }
    const v16h bh = __builtin_bit_cast(v16h, bh_pk);

    // gates + elementwise, grouped by tile parity to cap live accumulators
    #pragma unroll
    for (int mm = 0; mm < 2; ++mm) {
      v8f di = {}, df = {}, dg = {}, dO = {};
      di = __builtin_amdgcn_wmma_f32_16x16x32_f16(false, a_whh[0 + mm], false, bh, (short)0, di, false, false);
      di = __builtin_amdgcn_wmma_f32_16x16x32_f16(false, a_g[0 + mm],   false, bo, (short)0, di, false, false);
      df = __builtin_amdgcn_wmma_f32_16x16x32_f16(false, a_whh[2 + mm], false, bh, (short)0, df, false, false);
      df = __builtin_amdgcn_wmma_f32_16x16x32_f16(false, a_g[2 + mm],   false, bo, (short)0, df, false, false);
      dg = __builtin_amdgcn_wmma_f32_16x16x32_f16(false, a_whh[4 + mm], false, bh, (short)0, dg, false, false);
      dg = __builtin_amdgcn_wmma_f32_16x16x32_f16(false, a_g[4 + mm],   false, bo, (short)0, dg, false, false);
      dO = __builtin_amdgcn_wmma_f32_16x16x32_f16(false, a_whh[6 + mm], false, bh, (short)0, dO, false, false);
      dO = __builtin_amdgcn_wmma_f32_16x16x32_f16(false, a_g[6 + mm],   false, bo, (short)0, dO, false, false);

      #pragma unroll
      for (int v = 0; v < 8; ++v) {
        const float iv = fast_sigmoid_pre(di[v]);   // rows pre-scaled by 0.5
        const float fv = fast_sigmoid_pre(df[v]);
        const float gv = fast_tanh(dg[v]);          // unscaled
        const float ov = fast_sigmoid_pre(dO[v]);
        const float cn = fmaf(fv, cst[mm][v], iv * gv);
        cst[mm][v] = cn;
        hst[mm][v] = ov * fast_tanh(cn);
      }
    }
  }

  // ---------------- FC head: out[b] = h_last @ fc_w^T + fc_b ----------------------
  float acc0 = 0.0f, acc1 = 0.0f;
  #pragma unroll
  for (int mm = 0; mm < 2; ++mm) {
    const int jb = 8 * halfsel + 16 * mm;          // this lane's hidden-j base
    const v4f* f0 = (const v4f*)(fc_w + jb);
    const v4f* f1 = (const v4f*)(fc_w + H_DIM + jb);
    const v4f f0a = f0[0], f0b = f0[1], f1a = f1[0], f1b = f1[1];
    #pragma unroll
    for (int v = 0; v < 4; ++v) {
      acc0 += hst[mm][v] * f0a[v] + hst[mm][4 + v] * f0b[v];
      acc1 += hst[mm][v] * f1a[v] + hst[mm][4 + v] * f1b[v];
    }
  }
  // partner (lane^16) holds the other 16 hidden indices of the same batch column
  acc0 += i2f(__builtin_amdgcn_ds_swizzle(f2i(acc0), SWZ_SWAPX16));
  acc1 += i2f(__builtin_amdgcn_ds_swizzle(f2i(acc1), SWZ_SWAPX16));

  if (halfsel == 0) {
    float2 r;
    r.x = acc0 + fc_b[0];
    r.y = acc1 + fc_b[1];
    *(float2*)(out + (size_t)(rowBase + lane16) * 2) = r;
  }
}

extern "C" void kernel_launch(void* const* d_in, const int* in_sizes, int n_in,
                              void* d_out, int out_size, void* d_ws, size_t ws_size,
                              hipStream_t stream) {
  (void)in_sizes; (void)n_in; (void)out_size; (void)d_ws; (void)ws_size;
  const int*   x    = (const int*)d_in[0];
  const float* emb  = (const float*)d_in[1];
  const float* W_ih = (const float*)d_in[2];
  const float* W_hh = (const float*)d_in[3];
  const float* b_ih = (const float*)d_in[4];
  const float* b_hh = (const float*)d_in[5];
  const float* fc_w = (const float*)d_in[6];
  const float* fc_b = (const float*)d_in[7];
  float* out = (float*)d_out;

  dim3 grid(B_TOT / ROWS_PER_BLOCK);   // 256 blocks
  dim3 block(NTHREADS);                // 128 threads = 4 waves
  lstm_fused<<<grid, block, 0, stream>>>(x, emb, W_ih, W_hh, b_ih, b_hh, fc_w, fc_b, out);
}
